// GDTEncoder_63522566308486
// MI455X (gfx1250) — matrix operand order repaired and checked by reading, not compiled, not run on hardware
//
#include <hip/hip_runtime.h>

#define N_NODES 50000
#define D       256
#define DFF     1024
#define NH      8
#define HDIM    32
#define DEG     16
#define MTILES  (N_NODES / 16)        // 3125 (odd)
#define MPAIRS  ((MTILES + 1) / 2)    // 1563
#define ALPHA_C 0.15f
#define SLOPE_C 0.2f

typedef __attribute__((ext_vector_type(16))) __bf16       v16bf;
typedef __attribute__((ext_vector_type(8)))  float        v8f;
typedef __attribute__((ext_vector_type(4)))  unsigned int u32x4;

union Frag {
    v16bf v;
    u32x4 q[2];
    unsigned short s[16];
};

__device__ __forceinline__ unsigned short f2bf(float f) {
    unsigned u = __builtin_bit_cast(unsigned, f);
    u += 0x7FFFu + ((u >> 16) & 1u);            // round-to-nearest-even
    return (unsigned short)(u >> 16);
}

// ---------------------------------------------------------------------------
// One-shot f32 -> bf16 conversion (weights; tiny, runs once per launch)
// ---------------------------------------------------------------------------
__global__ void cvt_bf16_kernel(const float* __restrict__ x,
                                unsigned short* __restrict__ y, int n) {
    const int i = blockIdx.x * 256 + threadIdx.x;
    if (i < n) y[i] = f2bf(x[i]);
}

// ---------------------------------------------------------------------------
// LN1: h = LN(ent_feat) ; store f32 (residual) + bf16 (WMMA A operand)
// ---------------------------------------------------------------------------
__global__ void ln1_kernel(const float* __restrict__ x,
                           const float* __restrict__ g,
                           const float* __restrict__ b,
                           float* __restrict__ yf,
                           unsigned short* __restrict__ ybf) {
    __shared__ float s1[256], s2[256];
    const int n = blockIdx.x, t = threadIdx.x;
    const float v = x[(size_t)n * D + t];
    s1[t] = v; s2[t] = v * v;
    __syncthreads();
    for (int off = 128; off > 0; off >>= 1) {
        if (t < off) { s1[t] += s1[t + off]; s2[t] += s2[t + off]; }
        __syncthreads();
    }
    const float mu  = s1[0] * (1.0f / D);
    const float var = s2[0] * (1.0f / D) - mu * mu;
    const float yv  = (v - mu) * rsqrtf(var + 1e-5f) * g[t] + b[t];
    yf[(size_t)n * D + t]  = yv;
    ybf[(size_t)n * D + t] = f2bf(yv);
}

// LN2 over (featN + h), bf16 output only (FFN GEMM A operand)
__global__ void ln2_kernel(const float* __restrict__ xa,
                           const float* __restrict__ xb,
                           const float* __restrict__ g,
                           const float* __restrict__ b,
                           unsigned short* __restrict__ ybf) {
    __shared__ float s1[256], s2[256];
    const int n = blockIdx.x, t = threadIdx.x;
    const size_t i = (size_t)n * D + t;
    const float v = xa[i] + xb[i];
    s1[t] = v; s2[t] = v * v;
    __syncthreads();
    for (int off = 128; off > 0; off >>= 1) {
        if (t < off) { s1[t] += s1[t + off]; s2[t] += s2[t + off]; }
        __syncthreads();
    }
    const float mu  = s1[0] * (1.0f / D);
    const float var = s2[0] * (1.0f / D) - mu * mu;
    ybf[i] = f2bf((v - mu) * rsqrtf(var + 1e-5f) * g[t] + b[t]);
}

// ---------------------------------------------------------------------------
// WMMA core: one wave computes a 2x2 grid of 16x16 f32 tiles
// (2 m-tiles x 2 o-tiles) of A(bf16)[M,K] @ Wbf(bf16)[O,K]^T.
// Fragment reuse: each A frag and each B frag feeds 2 WMMAs ->
// 8 b128 loads per 4 WMMA issues.
// A layout (16x32 bf16): lane l -> row l&15, K-groups {8*(l>>4), 8*(l>>4)+16}
// B layout (32x16 bf16): lane l -> col l&15, K-groups {8*(l>>4), 8*(l>>4)+16}
// ---------------------------------------------------------------------------
__device__ __forceinline__ void wmma_tile22(const unsigned short* __restrict__ Abf,
                                            const unsigned short* __restrict__ Wbf,
                                            int K, int mtile0, bool m1ok, int otile0,
                                            v8f& c00, v8f& c01, v8f& c10, v8f& c11) {
    const int lane = threadIdx.x & 31;
    const int hi   = lane >> 4;
    const int ml   = lane & 15;
    const unsigned short* arow0 = Abf + (size_t)(mtile0 * 16 + ml) * K + hi * 8;
    const unsigned short* arow1 = m1ok ? (arow0 + (size_t)16 * K) : arow0;
    const unsigned short* wrow0 = Wbf + (size_t)(otile0 * 16 + ml) * K + hi * 8;
    const unsigned short* wrow1 = wrow0 + (size_t)16 * K;
    const v8f z = {0.f, 0.f, 0.f, 0.f, 0.f, 0.f, 0.f, 0.f};
    c00 = z; c01 = z; c10 = z; c11 = z;
    for (int ks = 0; ks < K; ks += 32) {
        Frag a0, a1, b0, b1;
        a0.q[0] = *(const u32x4*)(arow0 + ks);
        a0.q[1] = *(const u32x4*)(arow0 + ks + 16);
        a1.q[0] = *(const u32x4*)(arow1 + ks);
        a1.q[1] = *(const u32x4*)(arow1 + ks + 16);
        b0.q[0] = *(const u32x4*)(wrow0 + ks);
        b0.q[1] = *(const u32x4*)(wrow0 + ks + 16);
        b1.q[0] = *(const u32x4*)(wrow1 + ks);
        b1.q[1] = *(const u32x4*)(wrow1 + ks + 16);
        c00 = __builtin_amdgcn_wmma_f32_16x16x32_bf16(false, a0.v, false, b0.v,
                                                      (short)0, c00, false, false);
        c01 = __builtin_amdgcn_wmma_f32_16x16x32_bf16(false, a0.v, false, b1.v,
                                                      (short)0, c01, false, false);
        c10 = __builtin_amdgcn_wmma_f32_16x16x32_bf16(false, a1.v, false, b0.v,
                                                      (short)0, c10, false, false);
        c11 = __builtin_amdgcn_wmma_f32_16x16x32_bf16(false, a1.v, false, b1.v,
                                                      (short)0, c11, false, false);
    }
}

// feat = h @ W_ent^T  (f32 out); block: 8 waves x 2 o-tiles = 16 o-tiles,
// each wave also covers 2 m-tiles.
__global__ void gemm_feat(const unsigned short* __restrict__ Abf,
                          const unsigned short* __restrict__ Wbf,
                          float* __restrict__ out) {
    const int mtile0 = blockIdx.x * 2;
    const bool m1ok  = (mtile0 + 1) < MTILES;
    const int otile0 = (blockIdx.y * 8 + (threadIdx.x >> 5)) * 2;
    v8f c00, c01, c10, c11;
    wmma_tile22(Abf, Wbf, D, mtile0, m1ok, otile0, c00, c01, c10, c11);
    const int lane = threadIdx.x & 31;
    const int row0 = mtile0 * 16 + ((lane >> 4) * 8);
    const int col  = otile0 * 16 + (lane & 15);
#pragma unroll
    for (int r = 0; r < 8; r++) {
        out[(size_t)(row0 + r) * D + col]      = c00[r];
        out[(size_t)(row0 + r) * D + col + 16] = c01[r];
        if (m1ok) {
            out[(size_t)(row0 + 16 + r) * D + col]      = c10[r];
            out[(size_t)(row0 + 16 + r) * D + col + 16] = c11[r];
        }
    }
}

// t1 = relu(x @ W1^T + b1)  (bf16 out)
__global__ void gemm_ffn1(const unsigned short* __restrict__ Abf,
                          const unsigned short* __restrict__ Wbf,
                          const float* __restrict__ bias,
                          unsigned short* __restrict__ outbf) {
    const int mtile0 = blockIdx.x * 2;
    const bool m1ok  = (mtile0 + 1) < MTILES;
    const int otile0 = (blockIdx.y * 8 + (threadIdx.x >> 5)) * 2;
    v8f c00, c01, c10, c11;
    wmma_tile22(Abf, Wbf, D, mtile0, m1ok, otile0, c00, c01, c10, c11);
    const int lane = threadIdx.x & 31;
    const int row0 = mtile0 * 16 + ((lane >> 4) * 8);
    const int col  = otile0 * 16 + (lane & 15);
    const float bv0 = bias[col], bv1 = bias[col + 16];
#pragma unroll
    for (int r = 0; r < 8; r++) {
        float v00 = c00[r] + bv0, v01 = c01[r] + bv1;
        outbf[(size_t)(row0 + r) * DFF + col]      = f2bf(v00 > 0.f ? v00 : 0.f);
        outbf[(size_t)(row0 + r) * DFF + col + 16] = f2bf(v01 > 0.f ? v01 : 0.f);
        if (m1ok) {
            float v10 = c10[r] + bv0, v11 = c11[r] + bv1;
            outbf[(size_t)(row0 + 16 + r) * DFF + col]      = f2bf(v10 > 0.f ? v10 : 0.f);
            outbf[(size_t)(row0 + 16 + r) * DFF + col + 16] = f2bf(v11 > 0.f ? v11 : 0.f);
        }
    }
}

// out = t1 @ W2^T + b2 + featN + h   (final output)
__global__ void gemm_ffn2(const unsigned short* __restrict__ Abf,
                          const unsigned short* __restrict__ Wbf,
                          const float* __restrict__ bias,
                          const float* __restrict__ featN,
                          const float* __restrict__ hbuf,
                          float* __restrict__ out) {
    const int mtile0 = blockIdx.x * 2;
    const bool m1ok  = (mtile0 + 1) < MTILES;
    const int otile0 = (blockIdx.y * 8 + (threadIdx.x >> 5)) * 2;
    v8f c00, c01, c10, c11;
    wmma_tile22(Abf, Wbf, DFF, mtile0, m1ok, otile0, c00, c01, c10, c11);
    const int lane = threadIdx.x & 31;
    const int row0 = mtile0 * 16 + ((lane >> 4) * 8);
    const int col  = otile0 * 16 + (lane & 15);
    const float bv0 = bias[col], bv1 = bias[col + 16];
#pragma unroll
    for (int r = 0; r < 8; r++) {
        const size_t i0 = (size_t)(row0 + r) * D + col;
        out[i0]      = c00[r] + bv0 + featN[i0] + hbuf[i0];
        out[i0 + 16] = c01[r] + bv1 + featN[i0 + 16] + hbuf[i0 + 16];
        if (m1ok) {
            const size_t i1 = (size_t)(row0 + 16 + r) * D + col;
            out[i1]      = c10[r] + bv0 + featN[i1] + hbuf[i1];
            out[i1 + 16] = c11[r] + bv1 + featN[i1 + 16] + hbuf[i1 + 16];
        }
    }
}

// ---------------------------------------------------------------------------
// eh/et per-head dots: eh[n,h] = feat[n,h,:] . attn_h[h,:]
// ---------------------------------------------------------------------------
__global__ void ehet_kernel(const float* __restrict__ feat,
                            const float* __restrict__ ah,
                            const float* __restrict__ at,
                            float* __restrict__ eh,
                            float* __restrict__ et) {
    const int t  = threadIdx.x;
    const int n  = blockIdx.x * 32 + (t >> 3);
    const int hh = t & 7;
    if (n >= N_NODES) return;
    const float* f  = feat + (size_t)n * D + hh * HDIM;
    const float* wh = ah + hh * HDIM;
    const float* wt = at + hh * HDIM;
    float sh = 0.f, st = 0.f;
#pragma unroll
    for (int d = 0; d < HDIM; d++) { const float v = f[d]; sh += v * wh[d]; st += v * wt[d]; }
    eh[n * NH + hh] = sh;
    et[n * NH + hh] = st;
}

// ---------------------------------------------------------------------------
// Edge softmax per (dst node, head): edges for node n are n*16..n*16+15.
// 128 threads/node: j = t&15 (edge), h = t>>4; 16-wide shuffle reductions.
// ---------------------------------------------------------------------------
__global__ void attn_kernel(const float* __restrict__ eh,
                            const float* __restrict__ et,
                            const int* __restrict__ src,
                            float* __restrict__ a) {
    const int n = blockIdx.x, t = threadIdx.x;
    const int j = t & 15, hh = t >> 4;
    const int e = n * DEG + j;
    const int s = src[e];
    float ev = eh[s * NH + hh] + et[n * NH + hh];
    ev = ev > 0.f ? ev : SLOPE_C * ev;
    float m = ev;
    for (int o = 8; o >= 1; o >>= 1) m = fmaxf(m, __shfl_xor(m, o, 16));
    const float ex = __expf(ev - m);
    float ssum = ex;
    for (int o = 8; o >= 1; o >>= 1) ssum += __shfl_xor(ssum, o, 16);
    a[(size_t)e * NH + hh] = ex / ssum;
}

// ---------------------------------------------------------------------------
// One PPR hop: fout[n] = (1-a)*sum_j alpha[n,j,h]*fin[src[n,j]] + a*feat0[n]
// 256 threads/node: t -> (head t>>5, dim t&31); src/alpha staged in LDS.
// ---------------------------------------------------------------------------
__global__ void hop_kernel(const float* __restrict__ fin,
                           const float* __restrict__ a,
                           const int* __restrict__ src,
                           const float* __restrict__ feat0,
                           float* __restrict__ fout) {
    __shared__ int   ssrc[DEG];
    __shared__ float sa[DEG * NH];
    const int n = blockIdx.x, t = threadIdx.x;
    if (t < DEG)      ssrc[t] = src[n * DEG + t];
    if (t < DEG * NH) sa[t]   = a[(size_t)n * DEG * NH + t];
    __syncthreads();
    const int hh = t >> 5;
    float acc = 0.f;
#pragma unroll
    for (int j = 0; j < DEG; j++)
        acc += sa[j * NH + hh] * fin[(size_t)ssrc[j] * D + t];
    const size_t i = (size_t)n * D + t;
    fout[i] = (1.0f - ALPHA_C) * acc + ALPHA_C * feat0[i];
}

// ---------------------------------------------------------------------------
extern "C" void kernel_launch(void* const* d_in, const int* in_sizes, int n_in,
                              void* d_out, int out_size, void* d_ws, size_t ws_size,
                              hipStream_t stream) {
    (void)in_sizes; (void)n_in; (void)out_size; (void)ws_size;
    const float* ent_feat = (const float*)d_in[0];
    const float* W_ent    = (const float*)d_in[1];
    const float* attn_h   = (const float*)d_in[2];
    const float* attn_t   = (const float*)d_in[3];
    const float* ln1_g    = (const float*)d_in[4];
    const float* ln1_b    = (const float*)d_in[5];
    const float* ln2_g    = (const float*)d_in[6];
    const float* ln2_b    = (const float*)d_in[7];
    const float* W1       = (const float*)d_in[8];
    const float* b1       = (const float*)d_in[9];
    const float* W2       = (const float*)d_in[10];
    const float* b2       = (const float*)d_in[11];
    const int*   src      = (const int*)d_in[12];
    // d_in[13] (dst) unused: edges are contiguous per destination node.

    char* wsb = (char*)d_ws;
    size_t off = 0;
    auto carve = [&](size_t bytes) -> char* {
        char* p = wsb + off;
        off += (bytes + 255) & ~(size_t)255;
        return p;
    };
    float*          h_f32   = (float*)carve((size_t)N_NODES * D * 4);
    unsigned short* h_bf    = (unsigned short*)carve((size_t)N_NODES * D * 2);
    float*          feat0   = (float*)carve((size_t)N_NODES * D * 4);
    float*          buf1    = (float*)carve((size_t)N_NODES * D * 4);
    float*          buf2    = (float*)carve((size_t)N_NODES * D * 4);
    float*          eh      = (float*)carve((size_t)N_NODES * NH * 4);
    float*          et      = (float*)carve((size_t)N_NODES * NH * 4);
    float*          alpha   = (float*)carve((size_t)N_NODES * DEG * NH * 4);
    unsigned short* x_bf    = (unsigned short*)carve((size_t)N_NODES * D * 2);
    unsigned short* t1_bf   = (unsigned short*)carve((size_t)N_NODES * DFF * 2);
    unsigned short* Went_bf = (unsigned short*)carve((size_t)D * D * 2);
    unsigned short* W1_bf   = (unsigned short*)carve((size_t)DFF * D * 2);
    unsigned short* W2_bf   = (unsigned short*)carve((size_t)D * DFF * 2);

    // 0) weights -> bf16 (once; ~590K elements total)
    cvt_bf16_kernel<<<(D * D + 255) / 256, 256, 0, stream>>>(W_ent, Went_bf, D * D);
    cvt_bf16_kernel<<<(DFF * D + 255) / 256, 256, 0, stream>>>(W1, W1_bf, DFF * D);
    cvt_bf16_kernel<<<(D * DFF + 255) / 256, 256, 0, stream>>>(W2, W2_bf, D * DFF);
    // 1) LN1
    ln1_kernel<<<N_NODES, 256, 0, stream>>>(ent_feat, ln1_g, ln1_b, h_f32, h_bf);
    // 2) feat = h @ W_ent^T  (WMMA bf16; 2x2 register tiles)
    gemm_feat<<<dim3(MPAIRS, 1), 256, 0, stream>>>(h_bf, Went_bf, feat0);
    // 3) attention logits per node/head
    ehet_kernel<<<(N_NODES + 31) / 32, 256, 0, stream>>>(feat0, attn_h, attn_t, eh, et);
    // 4) edge softmax (contiguous 16-edge segments)
    attn_kernel<<<N_NODES, 128, 0, stream>>>(eh, et, src, alpha);
    // 5) 5 PPR diffusion hops (ping-pong)
    hop_kernel<<<N_NODES, 256, 0, stream>>>(feat0, alpha, src, feat0, buf1);
    hop_kernel<<<N_NODES, 256, 0, stream>>>(buf1, alpha, src, feat0, buf2);
    hop_kernel<<<N_NODES, 256, 0, stream>>>(buf2, alpha, src, feat0, buf1);
    hop_kernel<<<N_NODES, 256, 0, stream>>>(buf1, alpha, src, feat0, buf2);
    hop_kernel<<<N_NODES, 256, 0, stream>>>(buf2, alpha, src, feat0, buf1);
    // 6) LN2 over rst = featN + h
    ln2_kernel<<<N_NODES, 256, 0, stream>>>(buf1, h_f32, ln2_g, ln2_b, x_bf);
    // 7) FFN GEMM1 + relu (WMMA bf16)
    gemm_ffn1<<<dim3(MPAIRS, 4), 256, 0, stream>>>(x_bf, W1_bf, b1, t1_bf);
    // 8) FFN GEMM2 + bias + residual (WMMA bf16) -> d_out
    gemm_ffn2<<<dim3(MPAIRS, 1), 256, 0, stream>>>(t1_bf, W2_bf, b2, buf1, h_f32,
                                                   (float*)d_out);
}
